// AFT_FULL_64424509440749
// MI455X (gfx1250) — matrix operand order repaired and checked by reading, not compiled, not run on hardware
//
#include <hip/hip_runtime.h>
#include <hip/hip_bf16.h>
#include <math.h>

#define D_MODEL 512
#define SEQ_N   1024
#define BATCH   64

typedef __attribute__((ext_vector_type(16))) __bf16 v16bf;
typedef __attribute__((ext_vector_type(8)))  float  v8f;
typedef __attribute__((ext_vector_type(2)))  float  v2f;
typedef __attribute__((ext_vector_type(2)))  __bf16 v2bf;

union Frag {
  v16bf v;
  unsigned int u[8];
};

// f32 -> bf16 (RNE) via native conversion.
__device__ __forceinline__ unsigned short f2bf(float f) {
  union { __bf16 b; unsigned short u; } c;
  c.b = (__bf16)f;
  return c.u;
}

// Packed pair convert: lowers to a single v_cvt_pk_bf16_f32, result is the
// packed 32-bit value directly (no half re-packing).
__device__ __forceinline__ unsigned int pk2bf(float a, float b) {
  union { v2bf v; unsigned int u; } c;
  v2f f; f.x = a; f.y = b;
  c.v = __builtin_convertvector(f, v2bf);
  return c.u;
}

__device__ __forceinline__ void cvt_store4(unsigned short* dst, float4 f) {
  *(unsigned int*)(dst)     = pk2bf(f.x, f.y);
  *(unsigned int*)(dst + 2) = pk2bf(f.z, f.w);
}

// Async global->LDS 16-byte copy (CDNA5 VGLOBAL async op, tracked by ASYNCcnt).
// Low 32 bits of a generic pointer to LDS are the group-segment byte offset.
__device__ __forceinline__ void async_copy16(const void* gsrc, void* lds_dst) {
  unsigned int loff = (unsigned int)(unsigned long long)lds_dst;
  asm volatile("global_load_async_to_lds_b128 %0, %1, off"
               :: "v"(loff), "v"(gsrc) : "memory");
}
#define WAIT_ASYNC0() asm volatile("s_wait_asynccnt 0x0" ::: "memory")

// A fragment: 16x32 bf16. LDS tile row-major [m][k], stride 40 ushorts (80B:
// 16B-aligned rows, and 20-dword stride -> conflict-free since 20*m mod 64 is
// distinct for m=0..15). Lanes 0-15 hold K 0-7/16-23, lanes 16-31 K 8-15/24-31.
__device__ __forceinline__ void load_a_frag(Frag& f, const unsigned short (*lds)[40],
                                            int mbase, int lane) {
  const int m = mbase + (lane & 15);
  const int half = lane >> 4;
  #pragma unroll
  for (int r = 0; r < 8; ++r) {
    const int k0 = ((r & 4) << 2) + half * 8 + ((r & 3) << 1);
    f.u[r] = *(const unsigned int*)&lds[m][k0];
  }
}

// B fragment: 32x16 bf16, stored transposed in LDS as [n][k] so K-pairs are
// contiguous. Lanes 0-15 hold K 0-7/16-23, lanes 16-31 K 8-15/24-31.
__device__ __forceinline__ void load_b_frag(Frag& f, const unsigned short (*lds)[40],
                                            int nbase, int lane) {
  const int n = nbase + (lane & 15);
  const int half = lane >> 4;
  #pragma unroll
  for (int r = 0; r < 8; ++r) {
    const int k0 = ((r & 4) << 2) + half * 8 + ((r & 3) << 1);
    f.u[r] = *(const unsigned int*)&lds[n][k0];
  }
}

#define WMMA_BF16(acc, a, b) \
  acc = __builtin_amdgcn_wmma_f32_16x16x32_bf16(false, (a).v, false, (b).v, \
                                                (short)0, acc, false, false)

// ---------------------------------------------------------------------------
// eb16[i,j] = bf16(exp(pos_bias[i,j]))
// ---------------------------------------------------------------------------
__global__ __launch_bounds__(256) void aft_eb(const float* __restrict__ pos_bias,
                                              unsigned short* __restrict__ eb16) {
  size_t i = ((size_t)blockIdx.x * 256 + threadIdx.x) * 4;
  float4 f = *(const float4*)&pos_bias[i];
  float4 e;
  e.x = __expf(f.x); e.y = __expf(f.y); e.z = __expf(f.z); e.w = __expf(f.w);
  cvt_store4(&eb16[i], e);
}

// ---------------------------------------------------------------------------
// sq = sigmoid(x @ Wq^T + bq).  M=BATCH*SEQ_N, N=D_MODEL, K=D_MODEL.
// Block tile 128x64, wave tile 32x32, K-chunk 32, double-buffered LDS.
// ---------------------------------------------------------------------------
__global__ __launch_bounds__(256) void aft_q_proj(
    const float* __restrict__ x, const float* __restrict__ Wq,
    const float* __restrict__ bq, float* __restrict__ sq) {
  __shared__ alignas(16) unsigned short ldsA[2][128][40];
  __shared__ alignas(16) unsigned short ldsB[2][64][40];
  const int t = threadIdx.x, lane = t & 31, wave = t >> 5;
  const int wm = wave >> 1, wn = wave & 1;
  const int m0 = blockIdx.x * 128, n0 = blockIdx.y * 64;

  float4 ra[4], rb[2];
  auto gload = [&](int kc) {
    #pragma unroll
    for (int i = 0; i < 4; ++i) {               // X tile 128x32 f32
      int fi = t + i * 256;
      int row = fi >> 3, c4 = (fi & 7) << 2;
      ra[i] = *(const float4*)&x[(size_t)(m0 + row) * D_MODEL + kc + c4];
    }
    #pragma unroll
    for (int i = 0; i < 2; ++i) {               // W tile 64x32 f32
      int fi = t + i * 256;
      int row = fi >> 3, c4 = (fi & 7) << 2;
      rb[i] = *(const float4*)&Wq[(size_t)(n0 + row) * D_MODEL + kc + c4];
    }
  };
  auto lstore = [&](int buf) {
    #pragma unroll
    for (int i = 0; i < 4; ++i) {
      int fi = t + i * 256;
      int row = fi >> 3, c4 = (fi & 7) << 2;
      cvt_store4(&ldsA[buf][row][c4], ra[i]);
    }
    #pragma unroll
    for (int i = 0; i < 2; ++i) {
      int fi = t + i * 256;
      int row = fi >> 3, c4 = (fi & 7) << 2;
      cvt_store4(&ldsB[buf][row][c4], rb[i]);
    }
  };

  v8f acc[2][2] = {};
  gload(0);
  lstore(0);
  __syncthreads();
  int cur = 0;
  for (int kc = 0; kc < D_MODEL; kc += 32) {
    const bool nxt = (kc + 32) < D_MODEL;
    if (nxt) gload(kc + 32);
    Frag a0, a1, b0, b1;
    load_a_frag(a0, ldsA[cur], wm * 32, lane);
    load_a_frag(a1, ldsA[cur], wm * 32 + 16, lane);
    load_b_frag(b0, ldsB[cur], wn * 32, lane);
    load_b_frag(b1, ldsB[cur], wn * 32 + 16, lane);
    WMMA_BF16(acc[0][0], a0, b0);
    WMMA_BF16(acc[0][1], a0, b1);
    WMMA_BF16(acc[1][0], a1, b0);
    WMMA_BF16(acc[1][1], a1, b1);
    if (nxt) {
      lstore(cur ^ 1);
      __syncthreads();
    }
    cur ^= 1;
  }
  const int half = lane >> 4, nl = lane & 15;
  #pragma unroll
  for (int mi = 0; mi < 2; ++mi)
    #pragma unroll
    for (int ni = 0; ni < 2; ++ni)
      #pragma unroll
      for (int r = 0; r < 8; ++r) {
        int gm = m0 + wm * 32 + mi * 16 + r + half * 8;
        int gn = n0 + wn * 32 + ni * 16 + nl;
        float q = acc[mi][ni][r] + bq[gn];
        sq[(size_t)gm * D_MODEL + gn] = 1.0f / (1.0f + __expf(-q));
      }
}

// ---------------------------------------------------------------------------
// Dual GEMM sharing A: k = x@Wk^T+bk, v = x@Wv^T+bv.
// Epilogue: ek = exp(k); stores the bf16 concat B-matrix for the attend phase:
//   ekvek[row][n]     = bf16(ek*v)
//   ekvek[row][512+n] = bf16(ek)
// ---------------------------------------------------------------------------
__global__ __launch_bounds__(256) void aft_kv_proj(
    const float* __restrict__ x,
    const float* __restrict__ Wk, const float* __restrict__ bk,
    const float* __restrict__ Wv, const float* __restrict__ bv,
    unsigned short* __restrict__ ekvek) {
  __shared__ alignas(16) unsigned short ldsA[2][128][40];
  __shared__ alignas(16) unsigned short ldsBk[2][64][40];
  __shared__ alignas(16) unsigned short ldsBv[2][64][40];
  const int t = threadIdx.x, lane = t & 31, wave = t >> 5;
  const int wm = wave >> 1, wn = wave & 1;
  const int m0 = blockIdx.x * 128, n0 = blockIdx.y * 64;

  float4 ra[4], rbk[2], rbv[2];
  auto gload = [&](int kc) {
    #pragma unroll
    for (int i = 0; i < 4; ++i) {
      int fi = t + i * 256;
      int row = fi >> 3, c4 = (fi & 7) << 2;
      ra[i] = *(const float4*)&x[(size_t)(m0 + row) * D_MODEL + kc + c4];
    }
    #pragma unroll
    for (int i = 0; i < 2; ++i) {
      int fi = t + i * 256;
      int row = fi >> 3, c4 = (fi & 7) << 2;
      rbk[i] = *(const float4*)&Wk[(size_t)(n0 + row) * D_MODEL + kc + c4];
      rbv[i] = *(const float4*)&Wv[(size_t)(n0 + row) * D_MODEL + kc + c4];
    }
  };
  auto lstore = [&](int buf) {
    #pragma unroll
    for (int i = 0; i < 4; ++i) {
      int fi = t + i * 256;
      int row = fi >> 3, c4 = (fi & 7) << 2;
      cvt_store4(&ldsA[buf][row][c4], ra[i]);
    }
    #pragma unroll
    for (int i = 0; i < 2; ++i) {
      int fi = t + i * 256;
      int row = fi >> 3, c4 = (fi & 7) << 2;
      cvt_store4(&ldsBk[buf][row][c4], rbk[i]);
      cvt_store4(&ldsBv[buf][row][c4], rbv[i]);
    }
  };

  v8f acck[2][2] = {}, accv[2][2] = {};
  gload(0);
  lstore(0);
  __syncthreads();
  int cur = 0;
  for (int kc = 0; kc < D_MODEL; kc += 32) {
    const bool nxt = (kc + 32) < D_MODEL;
    if (nxt) gload(kc + 32);
    Frag a0, a1, k0, k1, v0, v1;
    load_a_frag(a0, ldsA[cur], wm * 32, lane);
    load_a_frag(a1, ldsA[cur], wm * 32 + 16, lane);
    load_b_frag(k0, ldsBk[cur], wn * 32, lane);
    load_b_frag(k1, ldsBk[cur], wn * 32 + 16, lane);
    load_b_frag(v0, ldsBv[cur], wn * 32, lane);
    load_b_frag(v1, ldsBv[cur], wn * 32 + 16, lane);
    WMMA_BF16(acck[0][0], a0, k0);
    WMMA_BF16(acck[0][1], a0, k1);
    WMMA_BF16(acck[1][0], a1, k0);
    WMMA_BF16(acck[1][1], a1, k1);
    WMMA_BF16(accv[0][0], a0, v0);
    WMMA_BF16(accv[0][1], a0, v1);
    WMMA_BF16(accv[1][0], a1, v0);
    WMMA_BF16(accv[1][1], a1, v1);
    if (nxt) {
      lstore(cur ^ 1);
      __syncthreads();
    }
    cur ^= 1;
  }
  const int half = lane >> 4, nl = lane & 15;
  #pragma unroll
  for (int mi = 0; mi < 2; ++mi)
    #pragma unroll
    for (int ni = 0; ni < 2; ++ni)
      #pragma unroll
      for (int r = 0; r < 8; ++r) {
        int gm = m0 + wm * 32 + mi * 16 + r + half * 8;
        int gn = n0 + wn * 32 + ni * 16 + nl;
        float kk = acck[mi][ni][r] + bk[gn];
        float vv = accv[mi][ni][r] + bv[gn];
        float ek = __expf(kk);
        ekvek[(size_t)gm * 1024 + gn]       = f2bf(ek * vv);
        ekvek[(size_t)gm * 1024 + 512 + gn] = f2bf(ek);
      }
}

// ---------------------------------------------------------------------------
// Per-batch dual GEMM sharing A (eb tiles):
//   num = eb @ ekv, den = eb @ ek,  out = sq * num / den  (fused epilogue)
// eb A-tile is a straight bf16 copy -> async global->LDS (ASYNCcnt path).
// ---------------------------------------------------------------------------
__global__ __launch_bounds__(256) void aft_attend(
    const unsigned short* __restrict__ eb16,
    const unsigned short* __restrict__ ekvek,
    const float* __restrict__ sq,
    float* __restrict__ out) {
  __shared__ alignas(16) unsigned short ldsA[2][128][40];
  __shared__ alignas(16) unsigned short ldsN[2][64][40];   // ekv tile, [d][j]
  __shared__ alignas(16) unsigned short ldsD[2][64][40];   // ek  tile, [d][j]
  const int t = threadIdx.x, lane = t & 31, wave = t >> 5;
  const int wm = wave >> 1, wn = wave & 1;
  const int i0 = blockIdx.x * 128, d0 = blockIdx.y * 64;
  const int b = blockIdx.z;
  const unsigned short* Bsrc = ekvek + (size_t)b * SEQ_N * 1024;

  unsigned int rn[4], rd[4];
  auto asyncA = [&](int jc, int buf) {            // eb tile 128x32, 16B chunks
    #pragma unroll
    for (int i = 0; i < 2; ++i) {
      int ci = t + i * 256;                       // 0..511
      int row = ci >> 2, g = (ci & 3) << 3;       // g in {0,8,16,24} ushorts
      async_copy16(&eb16[(size_t)(i0 + row) * SEQ_N + jc + g], &ldsA[buf][row][g]);
    }
  };
  auto gloadB = [&](int jc) {                     // B tiles 32x64 bf16
    #pragma unroll
    for (int i = 0; i < 4; ++i) {
      int ei = t + i * 256;                       // uint index 0..1023
      int d2 = (ei & 31) << 1, j = ei >> 5;
      rn[i] = *(const unsigned int*)&Bsrc[(size_t)(jc + j) * 1024 + d0 + d2];
      rd[i] = *(const unsigned int*)&Bsrc[(size_t)(jc + j) * 1024 + 512 + d0 + d2];
    }
  };
  auto lstoreB = [&](int buf) {                   // LDS transpose -> [d][j]
    #pragma unroll
    for (int i = 0; i < 4; ++i) {
      int ei = t + i * 256;
      int d2 = (ei & 31) << 1, j = ei >> 5;
      ldsN[buf][d2][j]     = (unsigned short)(rn[i] & 0xFFFFu);
      ldsN[buf][d2 + 1][j] = (unsigned short)(rn[i] >> 16);
      ldsD[buf][d2][j]     = (unsigned short)(rd[i] & 0xFFFFu);
      ldsD[buf][d2 + 1][j] = (unsigned short)(rd[i] >> 16);
    }
  };

  v8f accN[2][2] = {}, accD[2][2] = {};
  asyncA(0, 0);
  gloadB(0);
  lstoreB(0);
  WAIT_ASYNC0();
  __syncthreads();
  int cur = 0;
  for (int jc = 0; jc < SEQ_N; jc += 32) {
    const bool nxt = (jc + 32) < SEQ_N;
    if (nxt) {
      asyncA(jc + 32, cur ^ 1);
      gloadB(jc + 32);
    }
    Frag a0, a1, bn0, bn1, bd0, bd1;
    load_a_frag(a0, ldsA[cur], wm * 32, lane);
    load_a_frag(a1, ldsA[cur], wm * 32 + 16, lane);
    load_b_frag(bn0, ldsN[cur], wn * 32, lane);
    load_b_frag(bn1, ldsN[cur], wn * 32 + 16, lane);
    load_b_frag(bd0, ldsD[cur], wn * 32, lane);
    load_b_frag(bd1, ldsD[cur], wn * 32 + 16, lane);
    WMMA_BF16(accN[0][0], a0, bn0);
    WMMA_BF16(accN[0][1], a0, bn1);
    WMMA_BF16(accN[1][0], a1, bn0);
    WMMA_BF16(accN[1][1], a1, bn1);
    WMMA_BF16(accD[0][0], a0, bd0);
    WMMA_BF16(accD[0][1], a0, bd1);
    WMMA_BF16(accD[1][0], a1, bd0);
    WMMA_BF16(accD[1][1], a1, bd1);
    if (nxt) {
      lstoreB(cur ^ 1);
      WAIT_ASYNC0();
      __syncthreads();
    }
    cur ^= 1;
  }
  const int half = lane >> 4, nl = lane & 15;
  #pragma unroll
  for (int mi = 0; mi < 2; ++mi)
    #pragma unroll
    for (int ni = 0; ni < 2; ++ni)
      #pragma unroll
      for (int r = 0; r < 8; ++r) {
        int gi = i0 + wm * 32 + mi * 16 + r + half * 8;
        int gd = d0 + wn * 32 + ni * 16 + nl;
        size_t idx = ((size_t)b * SEQ_N + gi) * D_MODEL + gd;
        out[idx] = sq[idx] * (accN[mi][ni][r] / accD[mi][ni][r]);
      }
}

// ---------------------------------------------------------------------------
extern "C" void kernel_launch(void* const* d_in, const int* in_sizes, int n_in,
                              void* d_out, int out_size, void* d_ws, size_t ws_size,
                              hipStream_t stream) {
  const float* x  = (const float*)d_in[0];
  const float* Wq = (const float*)d_in[1];
  const float* bq = (const float*)d_in[2];
  const float* Wk = (const float*)d_in[3];
  const float* bk = (const float*)d_in[4];
  const float* Wv = (const float*)d_in[5];
  const float* bv = (const float*)d_in[6];
  const float* pb = (const float*)d_in[7];
  float* out = (float*)d_out;

  // workspace layout:
  //   sq     : BATCH*SEQ_N*D_MODEL f32  = 128 MiB
  //   ekvek  : BATCH*SEQ_N*1024   bf16  = 128 MiB
  //   eb16   : SEQ_N*SEQ_N        bf16  =   2 MiB
  char* ws = (char*)d_ws;
  float* sq = (float*)ws;
  size_t off = (size_t)BATCH * SEQ_N * D_MODEL * sizeof(float);
  unsigned short* ekvek = (unsigned short*)(ws + off);
  off += (size_t)BATCH * SEQ_N * 1024 * sizeof(unsigned short);
  unsigned short* eb16 = (unsigned short*)(ws + off);

  dim3 blk(256);
  aft_eb<<<dim3(((size_t)SEQ_N * SEQ_N) / 1024), blk, 0, stream>>>(pb, eb16);
  aft_q_proj<<<dim3((BATCH * SEQ_N) / 128, D_MODEL / 64), blk, 0, stream>>>(x, Wq, bq, sq);
  aft_kv_proj<<<dim3((BATCH * SEQ_N) / 128, D_MODEL / 64), blk, 0, stream>>>(x, Wk, bk, Wv, bv, ekvek);
  aft_attend<<<dim3(SEQ_N / 128, D_MODEL / 64, BATCH), blk, 0, stream>>>(eb16, ekvek, sq, out);
}
